// RelFTAttention2D_326417515055
// MI455X (gfx1250) — matrix operand order repaired and checked by reading, not compiled, not run on hardware
//
#include <hip/hip_runtime.h>
#include <hip/hip_bf16.h>

// ---------------- problem constants (from reference) ----------------
#define B_   8
#define H_   128
#define W_   128
#define C_   96
#define HW_  (H_ * W_)
#define BHW_ (B_ * H_ * W_)
#define EPS_       1e-5f
#define FT_SMOOTH_ 1e-5f
#define WTAP_ (C_ * C_)          // 9216 weights per 3x3 tap
#define WALL_ (9 * WTAP_)        // 82944 per conv

typedef __attribute__((ext_vector_type(16))) __bf16 v16bf;
typedef __attribute__((ext_vector_type(8)))  float  v8f;

__device__ __forceinline__ float sigmoidf_(float x) {
    return 1.0f / (1.0f + __expf(-x));
}

__device__ __forceinline__ float ftanimoto_scalar(float tpl, float tpp, float tll) {
    float num = tpl + FT_SMOOTH_;
    float den = 0.0f;
    float a = 1.0f;
    #pragma unroll
    for (int d = 0; d < 5; ++d) {
        float bcoef = -(2.0f * a - 1.0f);
        den += 1.0f / (a * (tpp + tll) + bcoef * tpl + FT_SMOOTH_);
        a *= 2.0f;
    }
    return num * den * 0.2f;   // 1/FT_DEPTH
}

// =====================================================================
// Kernel 0: weight prep — convert f32 HWIO [3,3,Cin,Cout] to bf16
// [tap][Cout][Cin] so conv-time B tiles are contiguous async copies.
// =====================================================================
__global__ __launch_bounds__(256)
void wprep(const float* __restrict__ w, __bf16* __restrict__ wt)
{
    const int idx = blockIdx.x * 256 + threadIdx.x;
    if (idx >= WALL_) return;
    const int tap  = idx / WTAP_;
    const int r    = idx % WTAP_;
    const int cout = r / C_;
    const int cin  = r % C_;
    wt[idx] = (__bf16)w[((size_t)tap * C_ + cin) * C_ + cout];
}

// =====================================================================
// Kernel 1: 3x3 SAME conv (96->96) + BN + sigmoid, implicit GEMM via
// v_wmma_f32_16x16x32_bf16. 128 threads = 4 waves; each wave owns a
// 16-pixel x 96-cout tile (6 n-tiles). Per kh: wave-private 18x96 halo
// row. Per (kh,kw): async-copy the 96x96 bf16 tap tile into LDS. The 3
// K-steps are software-pipelined with double-buffered fragments and
// sched_barrier(0) so next-step ds_loads overlap current-step WMMAs
// (partial dscnt waits instead of full drains).
// =====================================================================
__global__ __launch_bounds__(128)
void conv_bn_sig_wmma(const float*  __restrict__ x,    // [B,H,W,C] f32
                      const __bf16* __restrict__ wt,   // [9][Cout][Cin] bf16
                      const float* __restrict__ gam,
                      const float* __restrict__ bet,
                      const float* __restrict__ mean,
                      const float* __restrict__ var,
                      float* __restrict__ out)         // [B,H,W,C] f32
{
    __shared__ __align__(16) __bf16 lA[4][18 * 96];   // per-wave halo row [pix][cin]
    __shared__ __align__(16) __bf16 lB[96 * 96];      // weight tap tile [cout][cin]

    const int tid  = threadIdx.x;
    const int lane = tid & 31;
    const int wave = tid >> 5;

    const int wpix0 = blockIdx.x * 64 + wave * 16;    // this wave's 16 pixels
    const int b   = wpix0 / HW_;
    const int rem = wpix0 % HW_;
    const int h   = rem / W_;
    const int w0  = rem % W_;                         // 16-aligned -> in-row group

    v8f zero = {};
    v8f acc[6];
    #pragma unroll
    for (int n = 0; n < 6; ++n) acc[n] = zero;

    const int m    = lane & 15;                 // A row / C-D column index
    const int kbA  = (lane < 16) ? 0 : 8;       // A fragment K base (elements)
    const int kbB  = (lane < 16) ? 0 : 8;       // B fragment K base (elements)
    const int ncol = lane & 15;                 // B column within n-tile

    union Frag { v16bf v; unsigned u[8]; };

    const unsigned ldsB0 = (unsigned)(size_t)&lB[0];  // LDS byte offset (addr[31:0])

    for (int kh = 0; kh < 3; ++kh) {
        const int  hy  = h + kh - 1;
        const bool hok = (hy >= 0) && (hy < H_);
        const int  hyc = hok ? hy : (hy < 0 ? 0 : H_ - 1);
        const float hmask = hok ? 1.0f : 0.0f;
        const float* xrow = x + ((size_t)b * H_ + hyc) * W_ * C_;

        // ---- stage wave-private 18x96 halo row (no barriers needed) ----
        #pragma unroll 6
        for (int i = 0; i < 18; ++i) {
            const int wx  = w0 + i - 1;
            const int wxc = (wx < 0) ? 0 : ((wx > W_ - 1) ? (W_ - 1) : wx);
            const float msk = (wx >= 0 && wx < W_) ? hmask : 0.0f;
            #pragma unroll
            for (int j = 0; j < 3; ++j) {
                const int c = lane + 32 * j;
                lA[wave][i * 96 + c] = (__bf16)(xrow[(size_t)wxc * C_ + c] * msk);
            }
        }

        // prefetch next kh's x row while we compute this one
        if (kh < 2) {
            const int hn = (h + kh >= 0 && h + kh < H_) ? (h + kh) : hyc;
            __builtin_prefetch(x + ((size_t)b * H_ + hn) * W_ * C_ + (size_t)w0 * C_ + lane, 0, 1);
        }

        for (int kw = 0; kw < 3; ++kw) {
            __syncthreads();   // previous tap tile fully consumed

            // ---- async-copy 96x96 bf16 tap tile (18432 B) into LDS ----
            {
                const __bf16* wtile = wt + (size_t)(kh * 3 + kw) * WTAP_;
                #pragma unroll 4
                for (int i = 0; i < 36; ++i) {
                    const int dw = tid + 128 * i;                 // dword index
                    const unsigned la = ldsB0 + (unsigned)(dw * 4);
                    const __bf16* ga = wtile + dw * 2;
                    asm volatile("global_load_async_to_lds_b32 %0, %1, off"
                                 :: "v"(la), "v"(ga) : "memory");
                }
                asm volatile("s_wait_asynccnt 0x0" ::: "memory");
            }
            __syncthreads();   // tap tile visible to all waves

            // ---- software-pipelined K-steps, double-buffered fragments ----
            Frag af[2];
            Frag bfs[2][6];

            auto load_frags = [&](int cb, int buf) {
                const unsigned* pa =
                    (const unsigned*)&lA[wave][(m + kw) * 96 + cb * 32];
                #pragma unroll
                for (int j = 0; j < 8; ++j) {
                    const int du = (kbA >> 1) + ((j < 4) ? j : (4 + j));
                    af[buf].u[j] = pa[du];
                }
                #pragma unroll
                for (int nt = 0; nt < 6; ++nt) {
                    const unsigned* pb = (const unsigned*)
                        &lB[(nt * 16 + ncol) * 96 + cb * 32 + kbB];
                    #pragma unroll
                    for (int j = 0; j < 8; ++j) bfs[buf][nt].u[j] = pb[(j < 4) ? j : (4 + j)];
                }
            };

            load_frags(0, 0);
            #pragma unroll
            for (int cb = 0; cb < 3; ++cb) {
                const int cur = cb & 1;
                if (cb < 2) load_frags(cb + 1, cur ^ 1);  // overlaps WMMAs below
                __builtin_amdgcn_sched_barrier(0);
                #pragma unroll
                for (int nt = 0; nt < 6; ++nt) {
                    acc[nt] = __builtin_amdgcn_wmma_f32_16x16x32_bf16(
                        false, af[cur].v, false, bfs[cur][nt].v, (short)0,
                        acc[nt], false, false);
                }
                __builtin_amdgcn_sched_barrier(0);
            }
        }
    }

    // ---- epilogue: BN + sigmoid, f32 16x16 C/D layout (VGPR r <-> row r/r+8) ----
    const int rowbase = (lane < 16) ? 0 : 8;
    #pragma unroll
    for (int nt = 0; nt < 6; ++nt) {
        const int ch = nt * 16 + ncol;
        const float sc = gam[ch] * rsqrtf(var[ch] + EPS_);
        const float mu = mean[ch];
        const float bb = bet[ch];
        #pragma unroll
        for (int r = 0; r < 8; ++r) {
            const int pix = wpix0 + rowbase + r;
            float y = (acc[nt][r] - mu) * sc + bb;
            out[(size_t)pix * C_ + ch] = sigmoidf_(y);
        }
    }
}

// =====================================================================
// Kernel 2a: per-(b,c) partial sums of q*k, q*q, k*k over 128-pixel
// chunks (deterministic two-phase reduction). part: [b][chunk][3][C]
// =====================================================================
__global__ __launch_bounds__(96)
void chan_partial(const float* __restrict__ q, const float* __restrict__ k,
                  float* __restrict__ part)
{
    const int c  = threadIdx.x;      // 0..95
    const int b  = blockIdx.y;       // 0..7
    const int ck = blockIdx.x;       // 0..127 (chunk)
    const int p0 = ck * 128;

    float tpl = 0.f, tpp = 0.f, tll = 0.f;
    for (int i = 0; i < 128; ++i) {
        const size_t idx = ((size_t)b * HW_ + p0 + i) * C_ + c;
        const float qv = q[idx];
        const float kv = k[idx];
        tpl += qv * kv;
        tpp += qv * qv;
        tll += kv * kv;
    }
    float* dst = part + (((size_t)b * 128 + ck) * 3) * C_ + c;
    dst[0]      = tpl;
    dst[C_]     = tpp;
    dst[2 * C_] = tll;
}

// Kernel 2b: finish channel attention: att_chan[b,c]
__global__ __launch_bounds__(256)
void chan_final(const float* __restrict__ part, float* __restrict__ attc)
{
    const int t = blockIdx.x * blockDim.x + threadIdx.x;
    if (t >= B_ * C_) return;
    const int b = t / C_;
    const int c = t % C_;
    float tpl = 0.f, tpp = 0.f, tll = 0.f;
    for (int j = 0; j < 128; ++j) {
        const float* pp = part + (((size_t)b * 128 + j) * 3) * C_ + c;
        tpl += pp[0];
        tpp += pp[C_];
        tll += pp[2 * C_];
    }
    attc[t] = ftanimoto_scalar(tpl, tpp, tll);
}

// =====================================================================
// Kernel 3: spatial attention: one wave32 per pixel, 3 coalesced channel
// reads per lane (96 = 3*32), shfl_xor tree reduction.
// =====================================================================
__global__ __launch_bounds__(256)
void spat_att(const float* __restrict__ q, const float* __restrict__ k,
              float* __restrict__ atts)
{
    const int lane = threadIdx.x & 31;
    const int wave = threadIdx.x >> 5;
    const size_t p = (size_t)blockIdx.x * 8 + wave;

    const float* qp = q + p * C_;
    const float* kp = k + p * C_;
    float tpl = 0.f, tpp = 0.f, tll = 0.f;
    #pragma unroll
    for (int j = 0; j < 3; ++j) {
        const float qv = qp[lane + 32 * j];
        const float kv = kp[lane + 32 * j];
        tpl += qv * kv;
        tpp += qv * qv;
        tll += kv * kv;
    }
    #pragma unroll
    for (int o = 16; o > 0; o >>= 1) {
        tpl += __shfl_xor(tpl, o, 32);
        tpp += __shfl_xor(tpp, o, 32);
        tll += __shfl_xor(tll, o, 32);
    }
    if (lane == 0)
        atts[p] = ftanimoto_scalar(tpl, tpp, tll);
}

// =====================================================================
// Kernel 4: out = BN( 0.5*(att_chan*v + att_spat*v) )
// =====================================================================
__global__ __launch_bounds__(256)
void finalize_k(const float* __restrict__ v, const float* __restrict__ atts,
                const float* __restrict__ attc,
                const float* __restrict__ gn, const float* __restrict__ bnb,
                const float* __restrict__ mn, const float* __restrict__ vn,
                float* __restrict__ out)
{
    const size_t t = (size_t)blockIdx.x * 256 + threadIdx.x;
    if (t >= (size_t)BHW_ * C_) return;
    const int    c = (int)(t % C_);
    const size_t p = t / C_;
    const int    b = (int)(p / HW_);

    const float vv = v[t];
    const float y  = 0.5f * (attc[b * C_ + c] * vv + atts[p] * vv);
    const float sc = gn[c] * rsqrtf(vn[c] + EPS_);
    out[t] = (y - mn[c]) * sc + bnb[c];
}

// =====================================================================
// Host launcher
// =====================================================================
extern "C" void kernel_launch(void* const* d_in, const int* in_sizes, int n_in,
                              void* d_out, int out_size, void* d_ws, size_t ws_size,
                              hipStream_t stream) {
    (void)in_sizes; (void)n_in; (void)out_size; (void)ws_size;

    const float* x1 = (const float*)d_in[0];
    const float* x2 = (const float*)d_in[1];
    const float* x3 = (const float*)d_in[2];
    const float* wq = (const float*)d_in[3];
    const float* gq = (const float*)d_in[4];
    const float* bq = (const float*)d_in[5];
    const float* mq = (const float*)d_in[6];
    const float* vq = (const float*)d_in[7];
    const float* wk = (const float*)d_in[8];
    const float* gk = (const float*)d_in[9];
    const float* bk = (const float*)d_in[10];
    const float* mk = (const float*)d_in[11];
    const float* vk = (const float*)d_in[12];
    const float* wv = (const float*)d_in[13];
    const float* gv = (const float*)d_in[14];
    const float* bv = (const float*)d_in[15];
    const float* mv = (const float*)d_in[16];
    const float* vv = (const float*)d_in[17];
    const float* gn  = (const float*)d_in[18];
    const float* bnb = (const float*)d_in[19];
    const float* mn  = (const float*)d_in[20];
    const float* vn  = (const float*)d_in[21];
    float* out = (float*)d_out;

    const size_t NQ = (size_t)BHW_ * C_;      // 12,582,912 floats per tensor
    float* wsf  = (float*)d_ws;
    float* Q    = wsf;
    float* K    = wsf + NQ;
    float* V    = wsf + 2 * NQ;
    float* ATTS = wsf + 3 * NQ;                         // BHW_
    float* PART = ATTS + BHW_;                          // 8*128*3*96
    float* ATTC = PART + (size_t)B_ * 128 * 3 * C_;     // B_*C_
    __bf16* WTQ = (__bf16*)(ATTC + B_ * C_);            // 82944 bf16 each
    __bf16* WTK = WTQ + WALL_;
    __bf16* WTV = WTK + WALL_;

    // --- weight prep: f32 HWIO -> bf16 [tap][cout][cin] ---
    dim3 wgrid(WALL_ / 256), wblk(256);
    wprep<<<wgrid, wblk, 0, stream>>>(wq, WTQ);
    wprep<<<wgrid, wblk, 0, stream>>>(wk, WTK);
    wprep<<<wgrid, wblk, 0, stream>>>(wv, WTV);

    // --- three convs + BN + sigmoid (WMMA implicit GEMM) ---
    dim3 cgrid(BHW_ / 64), cblk(128);
    conv_bn_sig_wmma<<<cgrid, cblk, 0, stream>>>(x1, WTQ, gq, bq, mq, vq, Q);
    conv_bn_sig_wmma<<<cgrid, cblk, 0, stream>>>(x2, WTK, gk, bk, mk, vk, K);
    conv_bn_sig_wmma<<<cgrid, cblk, 0, stream>>>(x3, WTV, gv, bv, mv, vv, V);

    // --- channel attention (two-phase deterministic reduction) ---
    chan_partial<<<dim3(128, B_), 96, 0, stream>>>(Q, K, PART);
    chan_final<<<dim3((B_ * C_ + 255) / 256), 256, 0, stream>>>(PART, ATTC);

    // --- spatial attention ---
    spat_att<<<dim3(BHW_ / 8), 256, 0, stream>>>(Q, K, ATTS);

    // --- combine + final BN ---
    finalize_k<<<dim3((unsigned)((NQ + 255) / 256)), 256, 0, stream>>>(
        V, ATTS, ATTC, gn, bnb, mn, vn, out);
}
